// MessagePassing_54099408060558
// MI455X (gfx1250) — matrix operand order repaired and checked by reading, not compiled, not run on hardware
//
#include <hip/hip_runtime.h>

typedef __bf16 bf16;
typedef __attribute__((ext_vector_type(16))) __bf16 v16bf;
typedef __attribute__((ext_vector_type(8)))  __bf16 v8bf;
typedef __attribute__((ext_vector_type(8)))  float  v8f;

#define LATENT  128
#define NNODES  50000
#define NEDGES  512000
#define ROWS    32      // rows (edges/nodes) per block
#define TB      128     // 4 waves of 32
#define LNPAD   132     // padded f32 LDS row stride
#define LN_EPS  1e-5f

// packed bf16 weight offsets inside d_ws (in bf16 elements)
#define WOFF_EW0 0
#define WOFF_EW1 49152
#define WOFF_EW2 65536
#define WOFF_EW3 81920
#define WOFF_NW0 98304
#define WOFF_NW1 131072
#define WOFF_NW2 147456
#define WOFF_NW3 163840

// ---------------------------------------------------------------------------
// Pack a row-major [K x 128] f32 weight matrix into WMMA B-fragment order:
// tile t = kt*8 + nt (32x16 tile), lane l holds column n = nt*16 + (l&15),
// 16 contiguous bf16 elements e -> K = kt*32 + (l>=16 ? 8 : 0) + (e&7) + (e>=8 ? 16 : 0).
// Each lane's 16 values are contiguous (32B) -> single v16bf load in GEMM.
// ---------------------------------------------------------------------------
__global__ void pack_weights(const float* __restrict__ W, bf16* __restrict__ out, int K) {
    int i = blockIdx.x * blockDim.x + threadIdx.x;
    if (i >= K * LATENT) return;
    int e    = i & 15;
    int l    = (i >> 4) & 31;
    int tile = i >> 9;
    int kt = tile >> 3;
    int nt = tile & 7;
    int k = kt * 32 + ((l >> 4) << 3) + (e & 7) + ((e >> 3) << 4);
    int n = nt * 16 + (l & 15);
    out[i] = (bf16)W[k * LATENT + n];
}

__global__ void zero_f32(float* __restrict__ p, int n) {
    int i = blockIdx.x * blockDim.x + threadIdx.x;
    if (i < n) p[i] = 0.0f;
}

// Load A fragment (16x32 bf16) from LDS row-major [row][KD]:
// lane 0-15: m=lane, elems 0-7 = K kbase+0..7,  8-15 = K kbase+16..23
// lane16-31: m=lane-16, elems 0-7 = K kbase+8..15, 8-15 = K kbase+24..31
__device__ __forceinline__ v16bf load_a_frag(const bf16* rowp, int kbase, int khalf) {
    v8bf lo = *(const v8bf*)(rowp + kbase + khalf * 8);
    v8bf hi = *(const v8bf*)(rowp + kbase + 16 + khalf * 8);
    return __builtin_shufflevector(lo, hi, 0,1,2,3,4,5,6,7,8,9,10,11,12,13,14,15);
}

// ---------------------------------------------------------------------------
// One MLP layer: D(32x128) = A(32xKD) * W(KDx128) + bias, optional ReLU.
// Block = 4 waves; wave (mt, nhalf) computes a 16x64 slab (4 accum tiles).
// Intermediate layers write bf16 to sOutBf; final layer writes f32 to sOutF32.
// ---------------------------------------------------------------------------
template<int KD>
__device__ __forceinline__ void mlp_layer(const bf16* sIn, const bf16* __restrict__ wPack,
                                          const float* __restrict__ bias,
                                          bf16* sOutBf, float* sOutF32, bool relu,
                                          int mt, int nhalf, int lane) {
    v8f c[4] = {};
    const int khalf = lane >> 4;
    const int nl    = lane & 15;
    const bf16* rowp = sIn + (mt * 16 + nl) * KD;   // A row for this lane (m = lane&15)

    for (int kt = 0; kt < KD / 32; ++kt) {
        v16bf a = load_a_frag(rowp, kt * 32, khalf);
#pragma unroll
        for (int j = 0; j < 4; ++j) {
            int nt = nhalf * 4 + j;
            v16bf b = *(const v16bf*)(wPack + ((size_t)(kt * 8 + nt) * 32 + lane) * 16);
            c[j] = __builtin_amdgcn_wmma_f32_16x16x32_bf16(
                       false, a, false, b, (short)0, c[j], false, false);
        }
    }
    // epilogue: C layout -> lane row m = r + khalf*8, col n = nl (within tile)
#pragma unroll
    for (int j = 0; j < 4; ++j) {
        int n  = (nhalf * 4 + j) * 16 + nl;
        float bv = bias[n];
#pragma unroll
        for (int r = 0; r < 8; ++r) {
            int m = mt * 16 + khalf * 8 + r;
            float v = c[j][r] + bv;
            if (relu) v = v > 0.0f ? v : 0.0f;
            if (sOutBf) sOutBf[m * LATENT + n] = (bf16)v;
            else        sOutF32[m * LNPAD + n] = v;
        }
    }
}

// ---------------------------------------------------------------------------
// Edge stage: gather [x[s], x[r], edge_attr] -> 4-layer MLP -> LN ->
// write edge_new + atomicAdd into agg[receiver].
// ---------------------------------------------------------------------------
__global__ __launch_bounds__(TB) void edge_kernel(
    const float* __restrict__ x, const float* __restrict__ ea,
    const int* __restrict__ eidx, const bf16* __restrict__ wp,
    const float* __restrict__ b0, const float* __restrict__ b1,
    const float* __restrict__ b2, const float* __restrict__ b3,
    const float* __restrict__ gam, const float* __restrict__ bet,
    float* __restrict__ out_edge, float* agg) {

    __shared__ bf16  sIn[ROWS * 384];
    __shared__ bf16  sAct[2][ROWS * LATENT];
    __shared__ float sOut[ROWS * LNPAD];
    __shared__ int   sRecv[ROWS];

    const int t  = threadIdx.x;
    const int e0 = blockIdx.x * ROWS;

    for (int r = 0; r < ROWS; ++r) {
        int e   = e0 + r;
        int snd = eidx[e];
        int rcv = eidx[NEDGES + e];
        if (t == 0) sRecv[r] = rcv;
        sIn[r * 384 +       t] = (bf16)x[(size_t)snd * LATENT + t];
        sIn[r * 384 + 128 + t] = (bf16)x[(size_t)rcv * LATENT + t];
        sIn[r * 384 + 256 + t] = (bf16)ea[(size_t)e * LATENT + t];
    }
    __syncthreads();

    const int wave = t >> 5, lane = t & 31;
    const int mt = wave & 1, nhalf = wave >> 1;

    mlp_layer<384>(sIn,     wp + WOFF_EW0, b0, sAct[0], nullptr, true,  mt, nhalf, lane);
    __syncthreads();
    mlp_layer<128>(sAct[0], wp + WOFF_EW1, b1, sAct[1], nullptr, true,  mt, nhalf, lane);
    __syncthreads();
    mlp_layer<128>(sAct[1], wp + WOFF_EW2, b2, sAct[0], nullptr, true,  mt, nhalf, lane);
    __syncthreads();
    mlp_layer<128>(sAct[0], wp + WOFF_EW3, b3, nullptr, sOut,    false, mt, nhalf, lane);
    __syncthreads();

    // LayerNorm: 4 threads per row
    int row = t >> 2, part = t & 3;
    float s = 0.0f, s2 = 0.0f;
    for (int k = part * 32; k < part * 32 + 32; ++k) {
        float v = sOut[row * LNPAD + k];
        s += v; s2 += v * v;
    }
    s  += __shfl_xor(s, 1, 32);  s  += __shfl_xor(s, 2, 32);
    s2 += __shfl_xor(s2, 1, 32); s2 += __shfl_xor(s2, 2, 32);
    float mu  = s * (1.0f / 128.0f);
    float var = s2 * (1.0f / 128.0f) - mu * mu;
    float inv = rsqrtf(var + LN_EPS);

    int e   = e0 + row;
    int rcv = sRecv[row];
    for (int k = part * 32; k < part * 32 + 32; ++k) {
        float v = (sOut[row * LNPAD + k] - mu) * inv * gam[k] + bet[k];
        out_edge[(size_t)e * LATENT + k] = v;
        atomicAdd(&agg[(size_t)rcv * LATENT + k], v);
    }
}

// ---------------------------------------------------------------------------
// Node stage: [x, agg] -> 4-layer MLP -> LN -> x_new (overwrites agg region).
// ---------------------------------------------------------------------------
__global__ __launch_bounds__(TB) void node_kernel(
    const float* __restrict__ x, const float* agg, const bf16* __restrict__ wp,
    const float* __restrict__ b0, const float* __restrict__ b1,
    const float* __restrict__ b2, const float* __restrict__ b3,
    const float* __restrict__ gam, const float* __restrict__ bet,
    float* out_x) {

    __shared__ bf16  sIn[ROWS * 256];
    __shared__ bf16  sAct[2][ROWS * LATENT];
    __shared__ float sOut[ROWS * LNPAD];

    const int t  = threadIdx.x;
    const int n0 = blockIdx.x * ROWS;

    for (int r = 0; r < ROWS; ++r) {
        int node = n0 + r;
        bool ok = node < NNODES;
        float xv = ok ? x[(size_t)node * LATENT + t]   : 0.0f;
        float av = ok ? agg[(size_t)node * LATENT + t] : 0.0f;
        sIn[r * 256 +       t] = (bf16)xv;
        sIn[r * 256 + 128 + t] = (bf16)av;
    }
    __syncthreads();

    const int wave = t >> 5, lane = t & 31;
    const int mt = wave & 1, nhalf = wave >> 1;

    mlp_layer<256>(sIn,     wp + WOFF_NW0, b0, sAct[0], nullptr, true,  mt, nhalf, lane);
    __syncthreads();
    mlp_layer<128>(sAct[0], wp + WOFF_NW1, b1, sAct[1], nullptr, true,  mt, nhalf, lane);
    __syncthreads();
    mlp_layer<128>(sAct[1], wp + WOFF_NW2, b2, sAct[0], nullptr, true,  mt, nhalf, lane);
    __syncthreads();
    mlp_layer<128>(sAct[0], wp + WOFF_NW3, b3, nullptr, sOut,    false, mt, nhalf, lane);
    __syncthreads();

    int row = t >> 2, part = t & 3;
    float s = 0.0f, s2 = 0.0f;
    for (int k = part * 32; k < part * 32 + 32; ++k) {
        float v = sOut[row * LNPAD + k];
        s += v; s2 += v * v;
    }
    s  += __shfl_xor(s, 1, 32);  s  += __shfl_xor(s, 2, 32);
    s2 += __shfl_xor(s2, 1, 32); s2 += __shfl_xor(s2, 2, 32);
    float mu  = s * (1.0f / 128.0f);
    float var = s2 * (1.0f / 128.0f) - mu * mu;
    float inv = rsqrtf(var + LN_EPS);

    int node = n0 + row;
    if (node < NNODES) {
        for (int k = part * 32; k < part * 32 + 32; ++k) {
            float v = (sOut[row * LNPAD + k] - mu) * inv * gam[k] + bet[k];
            out_x[(size_t)node * LATENT + k] = v;
        }
    }
}

// ---------------------------------------------------------------------------
// Launcher. d_in order (jax dict flatten, sorted keys; per MLP: W0..3, b0..3,
// beta, g): x, edge_attr, edge_index, edge{W0,W1,W2,W3,b0,b1,b2,b3,beta,g},
// node{W0,W1,W2,W3,b0,b1,b2,b3,beta,g}.
// d_out = [x_new (50000*128) | edge_new (512000*128)] f32.
// ---------------------------------------------------------------------------
extern "C" void kernel_launch(void* const* d_in, const int* in_sizes, int n_in,
                              void* d_out, int out_size, void* d_ws, size_t ws_size,
                              hipStream_t stream) {
    const float* x    = (const float*)d_in[0];
    const float* ea   = (const float*)d_in[1];
    const int*   eidx = (const int*)d_in[2];

    const float* eW0 = (const float*)d_in[3];
    const float* eW1 = (const float*)d_in[4];
    const float* eW2 = (const float*)d_in[5];
    const float* eW3 = (const float*)d_in[6];
    const float* eb0 = (const float*)d_in[7];
    const float* eb1 = (const float*)d_in[8];
    const float* eb2 = (const float*)d_in[9];
    const float* eb3 = (const float*)d_in[10];
    const float* ebeta = (const float*)d_in[11];
    const float* eg    = (const float*)d_in[12];

    const float* nW0 = (const float*)d_in[13];
    const float* nW1 = (const float*)d_in[14];
    const float* nW2 = (const float*)d_in[15];
    const float* nW3 = (const float*)d_in[16];
    const float* nb0 = (const float*)d_in[17];
    const float* nb1 = (const float*)d_in[18];
    const float* nb2 = (const float*)d_in[19];
    const float* nb3 = (const float*)d_in[20];
    const float* nbeta = (const float*)d_in[21];
    const float* ng    = (const float*)d_in[22];

    float* out_x = (float*)d_out;                           // also agg accumulator
    float* out_e = (float*)d_out + (size_t)NNODES * LATENT;
    bf16*  wp    = (bf16*)d_ws;

    // Pack weights to bf16 WMMA-B layout (tiny; ~360KB, lives in L2).
    pack_weights<<<(384 * 128 + 255) / 256, 256, 0, stream>>>(eW0, wp + WOFF_EW0, 384);
    pack_weights<<<(128 * 128 + 255) / 256, 256, 0, stream>>>(eW1, wp + WOFF_EW1, 128);
    pack_weights<<<(128 * 128 + 255) / 256, 256, 0, stream>>>(eW2, wp + WOFF_EW2, 128);
    pack_weights<<<(128 * 128 + 255) / 256, 256, 0, stream>>>(eW3, wp + WOFF_EW3, 128);
    pack_weights<<<(256 * 128 + 255) / 256, 256, 0, stream>>>(nW0, wp + WOFF_NW0, 256);
    pack_weights<<<(128 * 128 + 255) / 256, 256, 0, stream>>>(nW1, wp + WOFF_NW1, 128);
    pack_weights<<<(128 * 128 + 255) / 256, 256, 0, stream>>>(nW2, wp + WOFF_NW2, 128);
    pack_weights<<<(128 * 128 + 255) / 256, 256, 0, stream>>>(nW3, wp + WOFF_NW3, 128);

    // Zero the aggregation buffer (x_new region of d_out).
    zero_f32<<<(NNODES * LATENT + 255) / 256, 256, 0, stream>>>(out_x, NNODES * LATENT);

    // Edge MLP + scatter-add.
    edge_kernel<<<NEDGES / ROWS, TB, 0, stream>>>(
        x, ea, eidx, wp, eb0, eb1, eb2, eb3, eg, ebeta, out_e, out_x);

    // Node MLP (reads agg, overwrites with x_new).
    node_kernel<<<(NNODES + ROWS - 1) / ROWS, TB, 0, stream>>>(
        x, out_x, wp, nb0, nb1, nb2, nb3, ng, nbeta, out_x);
}
